// BahdanauAttention_50989851738500
// MI455X (gfx1250) — matrix operand order repaired and checked by reading, not compiled, not run on hardware
//
#include <hip/hip_runtime.h>
#include <hip/hip_bf16.h>
#include <math.h>

// ---------------------------------------------------------------------------
// Bahdanau attention, fused, bf16 WMMA on gfx1250 (wave32).
// ---------------------------------------------------------------------------

typedef __attribute__((ext_vector_type(16))) __bf16 v16bf;
typedef __attribute__((ext_vector_type(8)))  float  v8f;

#define UNITS   1024
#define BATCH   32
#define TLEN    2048
#define MTILE   64              // rows of flattened (b,t) per block
#define XSTRIDE (UNITS + 8)     // padded bf16 LDS row stride (2064B, 16B aligned)

struct Frag { uint4 lo, hi; };  // 32 bytes -> bit_cast to v16bf

__device__ __forceinline__ unsigned short f32_to_bf16(float f) {
    union { float f; unsigned int u; } c; c.f = f;
    unsigned int r = c.u + 0x7FFFu + ((c.u >> 16) & 1u);   // RNE
    return (unsigned short)(r >> 16);
}

// Branch-free tanh: prefer the CDNA5 V_TANH_F32 transcendental, else a
// 4-op exp2-based expansion (v_exp_f32 + v_rcp_f32, no exec-mask branches).
__device__ __forceinline__ float fast_tanh(float x) {
#if __has_builtin(__builtin_amdgcn_tanhf)
    return __builtin_amdgcn_tanhf(x);
#elif __has_builtin(__builtin_amdgcn_tanh_f32)
    return __builtin_amdgcn_tanh_f32(x);
#else
    float xc = fminf(fmaxf(x, -15.f), 15.f);        // exp2 arg stays finite
    float e  = __builtin_amdgcn_exp2f(xc * 2.885390081777927f);  // 2*log2(e)
    return (e - 1.f) * __builtin_amdgcn_rcpf(e + 1.f);
#endif
}

// --------------------------------------------------------------------------
// Kernel 0: W1 f32 -> bf16 (once per launch; 2MB result lives in L2 after)
// --------------------------------------------------------------------------
__global__ void k_convert_w1(const float* __restrict__ w1,
                             unsigned short* __restrict__ w1b) {
    int i = blockIdx.x * blockDim.x + threadIdx.x;
    if (i < UNITS * UNITS) w1b[i] = f32_to_bf16(w1[i]);
}

// --------------------------------------------------------------------------
// Kernel 1: hb[b][v] = W1_b[v] + W2_b[v] + hidden[b,:] . W2_w[v,:]
// --------------------------------------------------------------------------
__global__ void k_hidden_bias(const float* __restrict__ hidden,
                              const float* __restrict__ W2,
                              const float* __restrict__ W1b,
                              const float* __restrict__ W2b,
                              float* __restrict__ hb) {
    __shared__ float h[UNITS];
    const int b = blockIdx.y;
    for (int i = threadIdx.x; i < UNITS; i += blockDim.x)
        h[i] = hidden[b * UNITS + i];
    __syncthreads();
    const int v = blockIdx.x * blockDim.x + threadIdx.x;
    const float* wrow = W2 + (size_t)v * UNITS;
    float acc = W1b[v] + W2b[v];
    #pragma unroll 4
    for (int u = 0; u < UNITS; ++u) acc += h[u] * wrow[u];
    hb[b * UNITS + v] = acc;
}

// --------------------------------------------------------------------------
// Kernel 2: fused  score[b,t] = V_w . tanh( x[b,t,:] @ W1^T + hb[b,:] )
//   block = 256 thr (8 waves). M-tile = 64 rows in LDS (bf16, 132KB).
//   wave (w>>1) -> 16-row quarter; wave (w&1) -> 512-col N half.
//   Per 16-wide N chunk: 32 x v_wmma_f32_16x16x32_bf16 over K=1024,
//   then tanh + V-dot epilogue, butterfly-reduce across lanes,
//   LDS atomicAdd combines the two N-half waves per row.
// --------------------------------------------------------------------------
__global__ __launch_bounds__(256) void k_scores(
    const float* __restrict__ x,
    const unsigned short* __restrict__ w1b,
    const float* __restrict__ hb,
    const float* __restrict__ Vw,
    float* __restrict__ scores)
{
    __shared__ unsigned short xs[MTILE * XSTRIDE];
    __shared__ float sc[MTILE];

    const int tid  = threadIdx.x;
    const int row0 = blockIdx.x * MTILE;          // flat (b*TLEN + t) row
    const int b    = row0 / TLEN;                 // MTILE divides TLEN

    // stage x tile (64 x 1024) as bf16 into LDS, coalesced
    const float* xg = x + (size_t)row0 * UNITS;
    for (int i = tid; i < MTILE * UNITS; i += 256) {
        const int r = i >> 10, c = i & (UNITS - 1);
        xs[r * XSTRIDE + c] = f32_to_bf16(xg[(size_t)r * UNITS + c]);
    }
    if (tid < MTILE) sc[tid] = 0.f;
    __syncthreads();

    const int lane  = tid & 31;
    const int wave  = tid >> 5;
    const int mbase = (wave >> 1) * 16;           // row quarter of tile
    const int nbase = (wave & 1) * 512;           // column half of W1
    const int lhalf = lane >> 4;                  // lane half (0/1)

    // A-fragment source: lane -> row (lane&15); K runs [k0+8*lhalf, +8) and +16
    const unsigned short* arow =
        xs + (mbase + (lane & 15)) * XSTRIDE + lhalf * 8;
    const float* hbb = hb + b * UNITS;

    float part[8];
    #pragma unroll
    for (int r = 0; r < 8; ++r) part[r] = 0.f;

    for (int nc = 0; nc < 32; ++nc) {
        const int n = nbase + nc * 16 + (lane & 15);      // W1 row == att column
        const unsigned short* brow = w1b + (size_t)n * UNITS + lhalf * 8;
        const float hbv = hbb[n];
        const float vwv = Vw[n];

        v8f c = {};
        #pragma unroll 4
        for (int k0 = 0; k0 < UNITS; k0 += 32) {
            Frag af, bf;
            af.lo = *(const uint4*)(arow + k0);
            af.hi = *(const uint4*)(arow + k0 + 16);
            const uint4* bp = (const uint4*)(brow + k0);
            bf.lo = bp[0];
            bf.hi = bp[2];
            c = __builtin_amdgcn_wmma_f32_16x16x32_bf16(
                    false, __builtin_bit_cast(v16bf, af),
                    false, __builtin_bit_cast(v16bf, bf),
                    (short)0, c, false, false);
        }
        // epilogue: lane holds rows M = r + 8*lhalf (within wave tile) at col n
        #pragma unroll
        for (int r = 0; r < 8; ++r)
            part[r] += vwv * fast_tanh(c[r] + hbv);
    }

    // reduce across the 16 lanes of each half (columns), then combine waves
    #pragma unroll
    for (int r = 0; r < 8; ++r) {
        float v = part[r];
        v += __shfl_xor(v, 8, 16);
        v += __shfl_xor(v, 4, 16);
        v += __shfl_xor(v, 2, 16);
        v += __shfl_xor(v, 1, 16);
        part[r] = v;
    }
    if ((lane & 15) == 0) {
        #pragma unroll
        for (int r = 0; r < 8; ++r)
            atomicAdd(&sc[mbase + lhalf * 8 + r], part[r]);
    }
    __syncthreads();
    if (tid < MTILE) scores[(size_t)row0 + tid] = sc[tid];
}

// --------------------------------------------------------------------------
// Kernel 3: softmax over T per batch; writes attention_weights to d_out
// --------------------------------------------------------------------------
__global__ void k_softmax(const float* __restrict__ scores,
                          float* __restrict__ wts) {
    __shared__ float red[256];
    const int b = blockIdx.x, tid = threadIdx.x;
    const float* s = scores + b * TLEN;

    float m = -3.4e38f;
    for (int i = tid; i < TLEN; i += 256) m = fmaxf(m, s[i]);
    red[tid] = m; __syncthreads();
    for (int o = 128; o; o >>= 1) {
        if (tid < o) red[tid] = fmaxf(red[tid], red[tid + o]);
        __syncthreads();
    }
    const float smax = red[0];
    __syncthreads();

    float sum = 0.f;
    for (int i = tid; i < TLEN; i += 256) sum += __expf(s[i] - smax);
    red[tid] = sum; __syncthreads();
    for (int o = 128; o; o >>= 1) {
        if (tid < o) red[tid] += red[tid + o];
        __syncthreads();
    }
    const float inv = 1.f / red[0];
    for (int i = tid; i < TLEN; i += 256)
        wts[b * TLEN + i] = __expf(s[i] - smax) * inv;
}

// --------------------------------------------------------------------------
// Kernel 4: context[b,u] = sum_t w[b,t] * x[b,t,u]   (bandwidth bound)
// --------------------------------------------------------------------------
__global__ void k_context(const float* __restrict__ x,
                          const float* __restrict__ wts,
                          float* __restrict__ ctx) {
    __shared__ float w[TLEN];
    const int b = blockIdx.y, tid = threadIdx.x;
    for (int i = tid; i < TLEN; i += 256) w[i] = wts[b * TLEN + i];
    __syncthreads();
    const int u = blockIdx.x * 256 + tid;
    const float* xb = x + (size_t)b * TLEN * UNITS + u;
    float acc = 0.f;
    #pragma unroll 4
    for (int t = 0; t < TLEN; ++t) acc += w[t] * xb[(size_t)t * UNITS];
    ctx[b * UNITS + u] = acc;
}

// --------------------------------------------------------------------------
extern "C" void kernel_launch(void* const* d_in, const int* in_sizes, int n_in,
                              void* d_out, int out_size, void* d_ws, size_t ws_size,
                              hipStream_t stream) {
    const float* x      = (const float*)d_in[0];   // (32, 2048, 1024)
    const float* hidden = (const float*)d_in[1];   // (32, 1024)
    const float* W1w    = (const float*)d_in[2];   // (1024, 1024)
    const float* W1b    = (const float*)d_in[3];   // (1024,)
    const float* W2w    = (const float*)d_in[4];   // (1024, 1024)
    const float* W2b    = (const float*)d_in[5];   // (1024,)
    const float* Vw     = (const float*)d_in[6];   // (1, 1024)
    // d_in[7] = V_b : constant shift, cancels in softmax -> unused
    float* out = (float*)d_out;                    // [ctx 32768 | weights 65536]

    unsigned short* w1b = (unsigned short*)d_ws;                       // 2 MB
    float* hb     = (float*)((char*)d_ws + (size_t)UNITS * UNITS * 2); // 128 KB
    float* scores = hb + BATCH * UNITS;                                // 256 KB

    float* wts = out + BATCH * UNITS;   // attention_weights region
    float* ctx = out;                   // context_vector region

    k_convert_w1<<<(UNITS * UNITS) / 256, 256, 0, stream>>>(W1w, w1b);
    k_hidden_bias<<<dim3(UNITS / 256, BATCH), 256, 0, stream>>>(hidden, W2w, W1b, W2b, hb);
    k_scores<<<(BATCH * TLEN) / MTILE, 256, 0, stream>>>(x, w1b, hb, Vw, scores);
    k_softmax<<<BATCH, 256, 0, stream>>>(scores, wts);
    k_context<<<dim3(UNITS / 256, BATCH), 256, 0, stream>>>(x, wts, ctx);
}
